// ClassifierHead_41025527611502
// MI455X (gfx1250) — matrix operand order repaired and checked by reading, not compiled, not run on hardware
//
#include <hip/hip_runtime.h>
#include <hip/hip_bf16.h>
#include <math.h>

typedef __attribute__((ext_vector_type(16))) __bf16 v16bf;
typedef __attribute__((ext_vector_type(8)))  float  v8f;

#define EPS_F        1e-15f
#define TANH_CLIP_F  (1.0f - 1e-5f)
// C = 1.0 so sqrt(C) = 1; TEMP = 0.1

__device__ __forceinline__ unsigned short f2bf_rne(float f) {
    unsigned int u = __float_as_uint(f);
    unsigned int r = u + 0x7fffu + ((u >> 16) & 1u);
    return (unsigned short)(r >> 16);
}

// ---------------------------------------------------------------------------
// Tiled WMMA GEMM:  C[M,N] = A[M,K](bf16) * Bt[N,K](bf16)^T  (+ epilogue)
// EPI 0: +bias, exact GELU, store bf16
// EPI 1: +bias, store f32
// EPI 2: hyperbolic pairwise-distance epilogue -> logits f32 (col-guarded)
// Block: 128 threads (4 waves); block tile 128x128; wave tile 64x64.
// ---------------------------------------------------------------------------
template <int EPI>
__global__ __launch_bounds__(128)
void gemm_wmma_bf16(const unsigned short* __restrict__ A,
                    const unsigned short* __restrict__ Bt,
                    const float* __restrict__ bias,
                    const float* __restrict__ A2,   // EPI2: proto sq-norms [N]
                    const float* __restrict__ B2,   // EPI2: row  sq-norms [M]
                    void* __restrict__ Out,
                    int M, int N, int K)
{
    constexpr int LDA = 40;   // bf16 elems per LDS row (padded vs 32)
    constexpr int LDB = 40;
    __shared__ unsigned short ldsA[128 * LDA];
    __shared__ unsigned short ldsB[128 * LDB];

    const int t     = threadIdx.x;
    const int lane  = t & 31;
    const int w     = t >> 5;
    const int half  = lane >> 4;     // 0 or 1
    const int l16   = lane & 15;
    const int waveM = (w >> 1) * 64;
    const int waveN = (w &  1) * 64;
    const int blockN = blockIdx.x * 128;
    const int blockM = blockIdx.y * 128;

    v8f acc[4][4];
    #pragma unroll
    for (int i = 0; i < 4; ++i)
        #pragma unroll
        for (int j = 0; j < 4; ++j) { v8f z = {}; acc[i][j] = z; }

    const int rr = t >> 2;           // 0..31: row within 32-row stripe
    const int cg = (t & 3) * 8;      // 0,8,16,24: k sub-column (8 bf16 = 16B)

    for (int k0 = 0; k0 < K; k0 += 32) {
        // ---- stage A tile [128 x 32] (rows always in range: M % 128 == 0)
        #pragma unroll
        for (int it = 0; it < 4; ++it) {
            const int r = rr + it * 32;
            const uint4 va = *(const uint4*)(A + (size_t)(blockM + r) * K + k0 + cg);
            *(uint4*)&ldsA[r * LDA + cg] = va;
        }
        // ---- stage Bt tile [128 x 32] (guard N: logits GEMM has N=1000)
        #pragma unroll
        for (int it = 0; it < 4; ++it) {
            const int r  = rr + it * 32;
            const int gn = blockN + r;
            uint4 vb = {0u, 0u, 0u, 0u};
            if (gn < N) vb = *(const uint4*)(Bt + (size_t)gn * K + k0 + cg);
            *(uint4*)&ldsB[r * LDB + cg] = vb;
        }
        __syncthreads();

        // ---- fragments (contiguous k order, same for A and B; not the ISA table order):
        // A 16x32: lane(l16,half) holds row m, k = 16*half+0..15 (contiguous)
        // B 32x16: lane(l16,half) holds col n, k = 16*half+0..15 (contiguous)
        union Frag { uint4 u[2]; v16bf v; };
        Frag af[4], bfr[4];
        #pragma unroll
        for (int ni = 0; ni < 4; ++ni) {
            const int n0 = waveN + ni * 16 + l16;
            bfr[ni].u[0] = *(const uint4*)&ldsB[n0 * LDB + 16 * half];
            bfr[ni].u[1] = *(const uint4*)&ldsB[n0 * LDB + 16 * half + 8];
        }
        #pragma unroll
        for (int mi = 0; mi < 4; ++mi) {
            const int m0 = waveM + mi * 16 + l16;
            af[mi].u[0] = *(const uint4*)&ldsA[m0 * LDA + 16 * half];
            af[mi].u[1] = *(const uint4*)&ldsA[m0 * LDA + 16 * half + 8];
        }
        #pragma unroll
        for (int mi = 0; mi < 4; ++mi)
            #pragma unroll
            for (int ni = 0; ni < 4; ++ni)
                acc[mi][ni] = __builtin_amdgcn_wmma_f32_16x16x32_bf16(
                    false, af[mi].v, false, bfr[ni].v,
                    (short)0, acc[mi][ni], false, false);
        __syncthreads();
    }

    // ---- epilogue; C/D layout: lane vgpr j -> m = 8*half + j, n = l16
    #pragma unroll
    for (int mi = 0; mi < 4; ++mi) {
        #pragma unroll
        for (int ni = 0; ni < 4; ++ni) {
            const int col = blockN + waveN + ni * 16 + l16;
            #pragma unroll
            for (int j = 0; j < 8; ++j) {
                const int row = blockM + waveM + mi * 16 + half * 8 + j;
                float v = acc[mi][ni][j];
                if (EPI == 0) {
                    v += bias[col];
                    v = 0.5f * v * (1.0f + erff(v * 0.70710678118654752f));
                    ((unsigned short*)Out)[(size_t)row * N + col] = f2bf_rne(v);
                } else if (EPI == 1) {
                    v += bias[col];
                    ((float*)Out)[(size_t)row * N + col] = v;
                } else {
                    if (col < N) {
                        const float ab = -v;               // reference: ab = -xh.proto
                        const float a2 = A2[col];
                        const float b2 = B2[row];
                        const float alpha = 1.0f + 2.0f * ab + b2;       // C = 1
                        const float beta  = 1.0f - a2;
                        const float gamma = 1.0f + 2.0f * ab + a2 * b2;
                        float m2 = (alpha * alpha * a2 + 2.0f * alpha * beta * ab
                                    + beta * beta * b2) / (gamma * gamma);
                        float mn  = sqrtf(fmaxf(m2, 0.0f));
                        float arg = fminf(mn, TANH_CLIP_F);
                        float dist = 2.0f * atanhf(arg);
                        ((float*)Out)[(size_t)row * N + col] = -dist * 10.0f; // /TEMP
                    }
                }
            }
        }
    }
}

// ---------------------------------------------------------------------------
__global__ __launch_bounds__(256)
void cvt_bf16_kernel(const float* __restrict__ in, unsigned short* __restrict__ out, int n)
{
    const int i = blockIdx.x * 256 + threadIdx.x;
    if (i < n) out[i] = f2bf_rne(in[i]);
}

// W[K,N] f32 -> Wt[N,K] bf16 (tiled transpose; K,N multiples of 32)
__global__ __launch_bounds__(256)
void transpose_cvt_kernel(const float* __restrict__ in, unsigned short* __restrict__ out,
                          int K, int N)
{
    __shared__ float tile[32][33];
    const int k0 = blockIdx.y * 32;
    const int n0 = blockIdx.x * 32;
    const int tx = threadIdx.x;   // 0..31
    const int ty = threadIdx.y;   // 0..7
    #pragma unroll
    for (int i = 0; i < 32; i += 8)
        tile[ty + i][tx] = in[(size_t)(k0 + ty + i) * N + n0 + tx];
    __syncthreads();
    #pragma unroll
    for (int i = 0; i < 32; i += 8)
        out[(size_t)(n0 + ty + i) * K + k0 + tx] = f2bf_rne(tile[tx][ty + i]);
}

// expmap0: u[rows,128] -> xh bf16, b2 = |xh|^2 per row. One wave per row.
__global__ __launch_bounds__(128)
void expmap_kernel(const float* __restrict__ u, unsigned short* __restrict__ xh,
                   float* __restrict__ b2out, int rows)
{
    const int w = threadIdx.x >> 5, lane = threadIdx.x & 31;
    const int row = blockIdx.x * 4 + w;
    if (row >= rows) return;
    const float* ur = u + (size_t)row * 128;
    float v[4]; float s = 0.0f;
    #pragma unroll
    for (int i = 0; i < 4; ++i) { v[i] = ur[lane + 32 * i]; s += v[i] * v[i]; }
    #pragma unroll
    for (int off = 16; off >= 1; off >>= 1) s += __shfl_xor(s, off, 32);
    const float n = fmaxf(sqrtf(s), EPS_F);
    const float scale = tanhf(n) / n;                // sqrt(C) = 1
    #pragma unroll
    for (int i = 0; i < 4; ++i)
        xh[(size_t)row * 128 + lane + 32 * i] = f2bf_rne(v[i] * scale);
    if (lane == 0) b2out[row] = scale * scale * s;
}

// prototypes: f32 -> bf16 + squared norms. One wave per row.
__global__ __launch_bounds__(128)
void proto_prep_kernel(const float* __restrict__ proto, unsigned short* __restrict__ pbf,
                       float* __restrict__ a2, int rows)
{
    const int w = threadIdx.x >> 5, lane = threadIdx.x & 31;
    const int row = blockIdx.x * 4 + w;
    if (row >= rows) return;
    const float* pr = proto + (size_t)row * 128;
    float v[4]; float s = 0.0f;
    #pragma unroll
    for (int i = 0; i < 4; ++i) { v[i] = pr[lane + 32 * i]; s += v[i] * v[i]; }
    #pragma unroll
    for (int off = 16; off >= 1; off >>= 1) s += __shfl_xor(s, off, 32);
    #pragma unroll
    for (int i = 0; i < 4; ++i)
        pbf[(size_t)row * 128 + lane + 32 * i] = f2bf_rne(v[i]);
    if (lane == 0) a2[row] = s;
}

// argmax per row, first-max tie-break (matches jnp.argmax).
__global__ __launch_bounds__(256)
void argmax_kernel(const float* __restrict__ logits, int* __restrict__ pred, int Ncols)
{
    __shared__ float sv[256];
    __shared__ int   si[256];
    const int b = blockIdx.x, t = threadIdx.x;
    float best = -3.4e38f; int bi = 0x7fffffff;
    for (int c = t; c < Ncols; c += 256) {
        const float v = logits[(size_t)b * Ncols + c];
        if (v > best) { best = v; bi = c; }
    }
    sv[t] = best; si[t] = bi;
    __syncthreads();
    for (int s = 128; s > 0; s >>= 1) {
        if (t < s) {
            const float ov = sv[t + s]; const int oi = si[t + s];
            if (ov > sv[t] || (ov == sv[t] && oi < si[t])) { sv[t] = ov; si[t] = oi; }
        }
        __syncthreads();
    }
    if (t == 0) pred[b] = si[0];
}

// feedback = logmap0(prototypes[pred]). One wave per row.
__global__ __launch_bounds__(128)
void feedback_kernel(const float* __restrict__ proto, const int* __restrict__ pred,
                     float* __restrict__ out, int rows)
{
    const int w = threadIdx.x >> 5, lane = threadIdx.x & 31;
    const int row = blockIdx.x * 4 + w;
    if (row >= rows) return;
    const int p = pred[row];
    const float* y = proto + (size_t)p * 128;
    float v[4]; float s = 0.0f;
    #pragma unroll
    for (int i = 0; i < 4; ++i) { v[i] = y[lane + 32 * i]; s += v[i] * v[i]; }
    #pragma unroll
    for (int off = 16; off >= 1; off >>= 1) s += __shfl_xor(s, off, 32);
    const float n = fmaxf(sqrtf(s), EPS_F);
    const float arg = fminf(n, TANH_CLIP_F);
    const float val = atanhf(arg) / n;
    #pragma unroll
    for (int i = 0; i < 4; ++i)
        out[(size_t)row * 128 + lane + 32 * i] = v[i] * val;
}

// ---------------------------------------------------------------------------
extern "C" void kernel_launch(void* const* d_in, const int* in_sizes, int n_in,
                              void* d_out, int out_size, void* d_ws, size_t ws_size,
                              hipStream_t stream)
{
    const float* x     = (const float*)d_in[0];
    const float* W1    = (const float*)d_in[1];
    const float* b1    = (const float*)d_in[2];
    const float* W2    = (const float*)d_in[3];
    const float* b2v   = (const float*)d_in[4];
    const float* W3    = (const float*)d_in[5];
    const float* b3    = (const float*)d_in[6];
    const float* proto = (const float*)d_in[7];

    const int B = 2048, Din = 1024, H = 4096, D = 128, P = 1000;

    char* ws = (char*)d_ws;
    size_t off = 0;
    auto carve = [&](size_t bytes) -> char* {
        char* p = ws + off;
        off = (off + bytes + 255) & ~(size_t)255;
        return p;
    };
    unsigned short* xbf = (unsigned short*)carve((size_t)B * Din * 2);
    unsigned short* W1t = (unsigned short*)carve((size_t)H * Din * 2);
    unsigned short* W2t = (unsigned short*)carve((size_t)H * H * 2);
    unsigned short* W3t = (unsigned short*)carve((size_t)D * H * 2);
    unsigned short* h1  = (unsigned short*)carve((size_t)B * H * 2);
    unsigned short* h2  = (unsigned short*)carve((size_t)B * H * 2);
    float*          u   = (float*)carve((size_t)B * D * 4);
    unsigned short* xh  = (unsigned short*)carve((size_t)B * D * 2);
    float*          b2r = (float*)carve((size_t)B * 4);
    unsigned short* pbf = (unsigned short*)carve((size_t)P * D * 2);
    float*          a2  = (float*)carve((size_t)P * 4);
    int*            prd = (int*)carve((size_t)B * 4);

    float* logits = (float*)d_out;                 // [B,P]
    float* fb     = (float*)d_out + (size_t)B * P; // [B,D]

    // One-time (per launch) precision/layout prep
    cvt_bf16_kernel<<<(B * Din + 255) / 256, 256, 0, stream>>>(x, xbf, B * Din);
    transpose_cvt_kernel<<<dim3(H / 32, Din / 32), dim3(32, 8), 0, stream>>>(W1, W1t, Din, H);
    transpose_cvt_kernel<<<dim3(H / 32, H / 32),   dim3(32, 8), 0, stream>>>(W2, W2t, H, H);
    transpose_cvt_kernel<<<dim3(D / 32, H / 32),   dim3(32, 8), 0, stream>>>(W3, W3t, H, D);
    proto_prep_kernel<<<(P + 3) / 4, 128, 0, stream>>>(proto, pbf, a2, P);

    // MLP
    gemm_wmma_bf16<0><<<dim3(H / 128, B / 128), 128, 0, stream>>>(
        xbf, W1t, b1, nullptr, nullptr, h1, B, H, Din);
    gemm_wmma_bf16<0><<<dim3(H / 128, B / 128), 128, 0, stream>>>(
        h1, W2t, b2v, nullptr, nullptr, h2, B, H, H);
    gemm_wmma_bf16<1><<<dim3(1, B / 128), 128, 0, stream>>>(
        h2, W3t, b3, nullptr, nullptr, u, B, D, H);

    // Hyperbolic head
    expmap_kernel<<<B / 4, 128, 0, stream>>>(u, xh, b2r, B);
    gemm_wmma_bf16<2><<<dim3((P + 127) / 128, B / 128), 128, 0, stream>>>(
        xh, pbf, nullptr, a2, b2r, logits, B, P, D);
    argmax_kernel<<<B, 256, 0, stream>>>(logits, prd, P);
    feedback_kernel<<<B / 4, 128, 0, stream>>>(proto, prd, fb, B);
}
